// QLinear_61186104099451
// MI455X (gfx1250) — matrix operand order repaired and checked by reading, not compiled, not run on hardware
//
#include <hip/hip_runtime.h>
#include <stdint.h>

// out[256,8192] = scale * (x[256,8192]_f32 @ W[8192,8192]_int8 ^T) + bias
// f16 WMMA path. int8 -> f16 is exact (|i| <= 128, 10-bit mantissa); x -> f16
// is the only rounding step. Per-tensor scale folded into the epilogue.

typedef __attribute__((ext_vector_type(16))) _Float16 v16h;
typedef __attribute__((ext_vector_type(8)))  _Float16 v8h;
typedef __attribute__((ext_vector_type(2)))  _Float16 h2;
typedef __attribute__((ext_vector_type(8)))  float    v8f;
typedef __attribute__((ext_vector_type(4)))  float    v4f;
typedef __attribute__((ext_vector_type(4)))  unsigned v4u;

#define KD 8192   // in_features (contiguous dim of both x and W)
#define ND 8192   // out_features
#define MD 256    // batch
#define MT 4      // 16-row WMMA tiles per wave (64 rows)
#define NT 2      // 16-col WMMA tiles per wave (32 cols)

union HF { v16h v; v8h h[2]; h2 p[8]; };

// 4 int8 (one dword) -> 4 exact f16, in element order.
// u = b ^ 0x80 is i+128 as unsigned; 0x6400|u is f16(1024 + u); minus 1152 -> i.
__device__ __forceinline__ void cvt4_i8_f16(unsigned raw, h2& lo, h2& hi) {
    const h2 off = {(_Float16)1152.0f, (_Float16)1152.0f};
    unsigned u = raw ^ 0x80808080u;
    lo = __builtin_bit_cast(h2, __builtin_amdgcn_perm(0x64646464u, u, 0x04010400u)) - off;
    hi = __builtin_bit_cast(h2, __builtin_amdgcn_perm(0x64646464u, u, 0x04030402u)) - off;
}

// ---------------- pre-conversion kernels (fast path) ----------------

__global__ __launch_bounds__(256) void convert_x_f16(const float* __restrict__ x,
                                                     _Float16* __restrict__ xh) {
    long idx = ((long)blockIdx.x * 256 + threadIdx.x) * 8;
    v4f a0 = *(const v4f*)(x + idx);
    v4f a1 = *(const v4f*)(x + idx + 4);
    v8h o;
    #pragma unroll
    for (int j = 0; j < 4; ++j) { o[j] = (_Float16)a0[j]; o[4 + j] = (_Float16)a1[j]; }
    *(v8h*)(xh + idx) = o;
}

__global__ __launch_bounds__(256) void convert_w_f16(const signed char* __restrict__ wq,
                                                     _Float16* __restrict__ wh) {
    long idx = ((long)blockIdx.x * 256 + threadIdx.x) * 16;
    v4u raw = *(const v4u*)(wq + idx);
    HF f;
    #pragma unroll
    for (int d = 0; d < 4; ++d) cvt4_i8_f16(raw[d], f.p[2 * d], f.p[2 * d + 1]);
    *(v8h*)(wh + idx)     = f.h[0];
    *(v8h*)(wh + idx + 8) = f.h[1];
}

// ---------------- GEMM, pre-converted f16 operands ----------------
// wave tile 64x32 (MT x NT); block = 4 waves (2M x 2N) -> 128x64 per block.
// K loop unrolled 2x with ping-pong fragment sets (register double-buffering,
// no copies -> no WMMA->VALU WAR hazard nops).

__global__ __launch_bounds__(128) void qlinear_wmma_pre(
    const _Float16* __restrict__ xh, const _Float16* __restrict__ wh,
    const float* __restrict__ scale, const float* __restrict__ bias,
    float* __restrict__ out)
{
    const int lane = threadIdx.x & 31;
    const int wave = threadIdx.x >> 5;
    const int half = lane >> 4;
    const int l16  = lane & 15;

    const int mBase = blockIdx.y * 128 + (wave & 1) * 64;
    const int nBase = blockIdx.x * 64  + (wave >> 1) * 32;

    const _Float16* xr[MT];
    #pragma unroll
    for (int mt = 0; mt < MT; ++mt)
        xr[mt] = xh + (long)(mBase + mt * 16 + l16) * KD + half * 16;
    const _Float16* wr[NT];
    #pragma unroll
    for (int nt = 0; nt < NT; ++nt)
        wr[nt] = wh + (long)(nBase + nt * 16 + l16) * KD + half * 16;

    v8f acc[MT][NT] = {};

#define LOAD_AB(A, B, K)                                                 \
    do {                                                                 \
        _Pragma("unroll")                                                \
        for (int mt = 0; mt < MT; ++mt) {                                \
            A[mt].h[0] = *(const v8h*)(xr[mt] + (K));        /* K=half*16+0..7     */ \
            A[mt].h[1] = *(const v8h*)(xr[mt] + (K) + 8);    /* K=half*16+8..15    */ \
        }                                                                \
        _Pragma("unroll")                                                \
        for (int nt = 0; nt < NT; ++nt) {                                \
            B[nt].h[0] = *(const v8h*)(wr[nt] + (K));        /* K=half*16+0..7     */ \
            B[nt].h[1] = *(const v8h*)(wr[nt] + (K) + 8);    /* K=half*16+8..15    */ \
        }                                                                \
    } while (0)

#define MMA(A, B)                                                        \
    do {                                                                 \
        _Pragma("unroll")                                                \
        for (int mt = 0; mt < MT; ++mt)                                  \
            _Pragma("unroll")                                            \
            for (int nt = 0; nt < NT; ++nt)                              \
                acc[mt][nt] = __builtin_amdgcn_wmma_f32_16x16x32_f16(    \
                    false, A[mt].v, false, B[nt].v, (short)0,            \
                    acc[mt][nt], false, false);                          \
    } while (0)

    HF a0[MT], b0[NT], a1[MT], b1[NT];
    LOAD_AB(a0, b0, 0);
    for (int k0 = 0; k0 < KD; k0 += 64) {
        LOAD_AB(a1, b1, k0 + 32);
        MMA(a0, b0);
        if (k0 + 64 < KD) LOAD_AB(a0, b0, k0 + 64);
        MMA(a1, b1);
    }
#undef LOAD_AB
#undef MMA

    const float s = scale[0];
    #pragma unroll
    for (int mt = 0; mt < MT; ++mt) {
        #pragma unroll
        for (int nt = 0; nt < NT; ++nt) {
            const int n = nBase + nt * 16 + l16;
            const float bv = bias[n];
            float* orow = out + (long)(mBase + mt * 16 + half * 8) * ND + n;
            #pragma unroll
            for (int v = 0; v < 8; ++v)
                orow[(long)v * ND] = s * acc[mt][nt][v] + bv;
        }
    }
}

// ---------------- GEMM, inline conversion (fallback, no workspace) ----------------

__global__ __launch_bounds__(128) void qlinear_wmma_inline(
    const float* __restrict__ x, const signed char* __restrict__ wq,
    const float* __restrict__ scale, const float* __restrict__ bias,
    float* __restrict__ out)
{
    const int lane = threadIdx.x & 31;
    const int wave = threadIdx.x >> 5;
    const int half = lane >> 4;
    const int l16  = lane & 15;

    const int mBase = blockIdx.y * 128 + (wave & 1) * 64;
    const int nBase = blockIdx.x * 64  + (wave >> 1) * 32;

    const float* xr[MT];
    #pragma unroll
    for (int mt = 0; mt < MT; ++mt)
        xr[mt] = x + (long)(mBase + mt * 16 + l16) * KD + half * 16;
    const signed char* wr[NT];
    #pragma unroll
    for (int nt = 0; nt < NT; ++nt)
        wr[nt] = wq + (long)(nBase + nt * 16 + l16) * KD + half * 16;

    v8f acc[MT][NT] = {};

#define LOAD_RAW(RA, RB, K)                                              \
    do {                                                                 \
        _Pragma("unroll")                                                \
        for (int mt = 0; mt < MT; ++mt) {                                \
            RA[mt][0] = *(const v4f*)(xr[mt] + (K));                     \
            RA[mt][1] = *(const v4f*)(xr[mt] + (K) + 4);                 \
            RA[mt][2] = *(const v4f*)(xr[mt] + (K) + 8);                 \
            RA[mt][3] = *(const v4f*)(xr[mt] + (K) + 12);                \
        }                                                                \
        _Pragma("unroll")                                                \
        for (int nt = 0; nt < NT; ++nt)                                  \
            RB[nt] = *(const v4u*)(wr[nt] + (K));                        \
    } while (0)

#define CVT_MMA(RA, RB)                                                  \
    do {                                                                 \
        HF a[MT], b[NT];                                                 \
        _Pragma("unroll")                                                \
        for (int mt = 0; mt < MT; ++mt)                                  \
            _Pragma("unroll")                                            \
            for (int j = 0; j < 4; ++j) {                                \
                a[mt].v[j]      = (_Float16)RA[mt][0][j];                \
                a[mt].v[4 + j]  = (_Float16)RA[mt][1][j];                \
                a[mt].v[8 + j]  = (_Float16)RA[mt][2][j];                \
                a[mt].v[12 + j] = (_Float16)RA[mt][3][j];                \
            }                                                            \
        _Pragma("unroll")                                                \
        for (int nt = 0; nt < NT; ++nt)                                  \
            _Pragma("unroll")                                            \
            for (int d = 0; d < 4; ++d)                                  \
                cvt4_i8_f16(RB[nt][d], b[nt].p[2 * d], b[nt].p[2 * d + 1]); \
        _Pragma("unroll")                                                \
        for (int mt = 0; mt < MT; ++mt)                                  \
            _Pragma("unroll")                                            \
            for (int nt = 0; nt < NT; ++nt)                              \
                acc[mt][nt] = __builtin_amdgcn_wmma_f32_16x16x32_f16(    \
                    false, a[mt].v, false, b[nt].v, (short)0,            \
                    acc[mt][nt], false, false);                          \
    } while (0)

    v4f ra0[MT][4], ra1[MT][4];
    v4u rb0[NT], rb1[NT];
    LOAD_RAW(ra0, rb0, 0);
    for (int k0 = 0; k0 < KD; k0 += 64) {
        LOAD_RAW(ra1, rb1, k0 + 32);
        CVT_MMA(ra0, rb0);
        if (k0 + 64 < KD) LOAD_RAW(ra0, rb0, k0 + 64);
        CVT_MMA(ra1, rb1);
    }
#undef LOAD_RAW
#undef CVT_MMA

    const float s = scale[0];
    #pragma unroll
    for (int mt = 0; mt < MT; ++mt) {
        #pragma unroll
        for (int nt = 0; nt < NT; ++nt) {
            const int n = nBase + nt * 16 + l16;
            const float bv = bias[n];
            float* orow = out + (long)(mBase + mt * 16 + half * 8) * ND + n;
            #pragma unroll
            for (int v = 0; v < 8; ++v)
                orow[(long)v * ND] = s * acc[mt][nt][v] + bv;
        }
    }
}

extern "C" void kernel_launch(void* const* d_in, const int* in_sizes, int n_in,
                              void* d_out, int out_size, void* d_ws, size_t ws_size,
                              hipStream_t stream) {
    const float*       x     = (const float*)d_in[0];
    const signed char* w     = (const signed char*)d_in[1];
    const float*       scale = (const float*)d_in[2];
    const float*       bias  = (const float*)d_in[3];
    float*             out   = (float*)d_out;

    const size_t XB = (size_t)MD * KD * sizeof(_Float16);   //   4 MB
    const size_t WB = (size_t)ND * KD * sizeof(_Float16);   // 128 MB

    dim3 gblk(128);                       // 4 waves: 2(M) x 2(N)
    dim3 ggrid(ND / 64, MD / 128);        // 128 x 2 = 256 workgroups

    if (ws_size >= XB + WB) {
        _Float16* xh = (_Float16*)d_ws;
        _Float16* wh = (_Float16*)((char*)d_ws + XB);
        convert_x_f16<<<dim3((unsigned)((long)MD * KD / 8 / 256)), dim3(256), 0, stream>>>(x, xh);
        convert_w_f16<<<dim3((unsigned)((long)ND * KD / 16 / 256)), dim3(256), 0, stream>>>(w, wh);
        qlinear_wmma_pre<<<ggrid, gblk, 0, stream>>>(xh, wh, scale, bias, out);
    } else {
        qlinear_wmma_inline<<<ggrid, gblk, 0, stream>>>(x, w, scale, bias, out);
    }
}
